// Tmp_6356551598442
// MI455X (gfx1250) — matrix operand order repaired and checked by reading, not compiled, not run on hardware
//
#include <hip/hip_runtime.h>
#include <hip/hip_bf16.h>

typedef __bf16 bf16;
typedef __attribute__((ext_vector_type(16))) __bf16 v16bf;
typedef __attribute__((ext_vector_type(8)))  __bf16 v8bf;
typedef __attribute__((ext_vector_type(4)))  __bf16 v4bf;
typedef __attribute__((ext_vector_type(8)))  float  v8f;

#define EMBED   1024
#define NHEADS  16
#define HDIM    64
#define SEQ     2048
#define BATCH   2
#define WIN     32

// ---------------------------------------------------------------------------
// CDNA5 async global->LDS copy (ASYNCcnt path), with safe fallback.
// Builtin signature (from hipcc diagnostic): first param is
//   int __attribute__((vector_size(16))) __device__ *   (AS1 = global)
// so the LDS param is the same vector type in AS3.
// ---------------------------------------------------------------------------
#if defined(__HIP_DEVICE_COMPILE__) && __has_builtin(__builtin_amdgcn_global_load_async_to_lds_b128)
#define HAVE_ASYNC_LDS 1
typedef int v4i __attribute__((vector_size(16)));
typedef __attribute__((address_space(1))) v4i gv4i;
typedef __attribute__((address_space(3))) v4i lv4i;
#else
#define HAVE_ASYNC_LDS 0
#endif

__device__ __forceinline__ void copy16_g2l(const bf16* g, bf16* l) {
#if HAVE_ASYNC_LDS
    __builtin_amdgcn_global_load_async_to_lds_b128((gv4i*)g, (lv4i*)l, 0, 0);
#else
    *(v8bf*)l = *(const v8bf*)g;
#endif
}

__device__ __forceinline__ void wait_async_lds() {
#if HAVE_ASYNC_LDS
#if __has_builtin(__builtin_amdgcn_s_wait_asynccnt)
    __builtin_amdgcn_s_wait_asynccnt(0);
#endif
#endif
}

// ---------------------------------------------------------------------------
// fp32 -> bf16 conversion (vectorized x4)
// ---------------------------------------------------------------------------
__global__ void cvt_f32_bf16(const float* __restrict__ in, bf16* __restrict__ out, int n4) {
    int i = blockIdx.x * blockDim.x + threadIdx.x;
    if (i >= n4) return;
    const float4 f = ((const float4*)in)[i];
    v4bf o;
    o[0] = (bf16)f.x; o[1] = (bf16)f.y; o[2] = (bf16)f.z; o[3] = (bf16)f.w;
    *(v4bf*)(out + (size_t)i * 4) = o;
}

// ---------------------------------------------------------------------------
// C[M,N] = A[M,K] @ W[N,K]^T + bias[N]
// A, W bf16 row-major (both K-contiguous), fp32 accumulate via WMMA.
// Block: 256 thr = 8 waves; block tile 128x128; wave tile 32x64 (2x4 frags).
// Double-buffered LDS, async global->LDS staging overlapped with WMMA.
// ---------------------------------------------------------------------------
#define BM 128
#define BN 128
#define BK 32
#define LDS_STRIDE 40   // 32 halves + 8 pad -> 80B row stride, conflict-free b128

__launch_bounds__(256)
__global__ void gemm_bias_wmma(const bf16* __restrict__ A,
                               const bf16* __restrict__ W,
                               const float* __restrict__ bias,
                               float* __restrict__ outF,   // one of outF/outB is null
                               bf16* __restrict__ outB,
                               int M, int N, int K)
{
    __shared__ bf16 As[2][BM * LDS_STRIDE];   // 2 x 10240 B
    __shared__ bf16 Ws[2][BN * LDS_STRIDE];   // 2 x 10240 B

    const int tid  = threadIdx.x;
    const int lane = tid & 31;
    const int wave = tid >> 5;
    const int wm   = (wave >> 1) * 32;     // 0,32,64,96
    const int wn   = (wave & 1) * 64;      // 0,64
    const int bm   = blockIdx.y * BM;
    const int bn   = blockIdx.x * BN;

    const int half = lane >> 4;            // lane half (0/1)
    const int l15  = lane & 15;
    const int aKlo = half * 16;            // A frag: K{0-15} vs K{16-31}
    const int bK0  = half * 16;            // B frag: K{0-15} vs K{16-31}

    v8f acc[2][4] = {};

    const int nk = K / BK;

    // stage one K-tile (A: 128x32 halves, W: 128x32 halves); 4 x 16B per thread
    auto stage = [&](int kt, int buf) {
        const int kb = kt * BK;
        #pragma unroll
        for (int t = 0; t < 2; ++t) {
            int c = tid + t * 256;
            int row = c >> 2, kc = (c & 3) * 8;
            copy16_g2l(A + (size_t)(bm + row) * K + kb + kc,
                       &As[buf][row * LDS_STRIDE + kc]);
        }
        #pragma unroll
        for (int t = 0; t < 2; ++t) {
            int c = tid + t * 256;
            int row = c >> 2, kc = (c & 3) * 8;
            copy16_g2l(W + (size_t)(bn + row) * K + kb + kc,
                       &Ws[buf][row * LDS_STRIDE + kc]);
        }
    };

    stage(0, 0);
    wait_async_lds();
    __syncthreads();

    for (int kt = 0; kt < nk; ++kt) {
        const int cur = kt & 1;
        if (kt + 1 < nk) stage(kt + 1, cur ^ 1);   // overlap with compute

        v16bf afrag[2], bfrag[4];
        #pragma unroll
        for (int i = 0; i < 2; ++i) {
            int r = wm + i * 16 + l15;
            union { v16bf v; v8bf h[2]; } u;
            u.h[0] = *(const v8bf*)(&As[cur][r * LDS_STRIDE + aKlo]);
            u.h[1] = *(const v8bf*)(&As[cur][r * LDS_STRIDE + aKlo + 8]);
            afrag[i] = u.v;
        }
        #pragma unroll
        for (int j = 0; j < 4; ++j) {
            int r = wn + j * 16 + l15;
            union { v16bf v; v8bf h[2]; } u;
            u.h[0] = *(const v8bf*)(&Ws[cur][r * LDS_STRIDE + bK0]);
            u.h[1] = *(const v8bf*)(&Ws[cur][r * LDS_STRIDE + bK0 + 8]);
            bfrag[j] = u.v;
        }

        #pragma unroll
        for (int i = 0; i < 2; ++i)
            #pragma unroll
            for (int j = 0; j < 4; ++j)
                acc[i][j] = __builtin_amdgcn_wmma_f32_16x16x32_bf16(
                    false, afrag[i], false, bfrag[j],
                    (short)0, acc[i][j], false, false);

        wait_async_lds();
        __syncthreads();
    }

    // ---- epilogue: D layout lane l, vgpr r -> m = r + 8*half, n = l&15 ----
    const int row0 = bm + wm + half * 8;
    const int col0 = bn + wn + l15;
    if (outF) {
        #pragma unroll
        for (int i = 0; i < 2; ++i) {
            #pragma unroll
            for (int j = 0; j < 4; ++j) {
                const int n = col0 + j * 16;
                const float bv = bias[n];
                float* p = outF + (size_t)(row0 + i * 16) * N + n;
                #pragma unroll
                for (int r = 0; r < 8; ++r) { p[0] = acc[i][j][r] + bv; p += N; }
            }
        }
    } else {
        #pragma unroll
        for (int i = 0; i < 2; ++i) {
            #pragma unroll
            for (int j = 0; j < 4; ++j) {
                const int n = col0 + j * 16;
                const float bv = bias[n];
                bf16* p = outB + (size_t)(row0 + i * 16) * N + n;
                #pragma unroll
                for (int r = 0; r < 8; ++r) { p[0] = (bf16)(acc[i][j][r] + bv); p += N; }
            }
        }
    }
}

// ---------------------------------------------------------------------------
// Sliding-window attention: one wave per (b, h, s); lanes split D (d, d+32).
// Online softmax, fp32 math on bf16 q/k/v; writes bf16 attn for O-projection.
// ---------------------------------------------------------------------------
__launch_bounds__(256)
__global__ void swin_attn(const bf16* __restrict__ q,
                          const bf16* __restrict__ k,
                          const bf16* __restrict__ v,
                          bf16* __restrict__ attn)
{
    int wid = blockIdx.x * (blockDim.x >> 5) + (threadIdx.x >> 5);
    if (wid >= BATCH * SEQ * NHEADS) return;
    const int lane = threadIdx.x & 31;
    const int h = wid & (NHEADS - 1);
    const int s = (wid >> 4) & (SEQ - 1);
    const int b = wid >> 15;

    const size_t rowQ = (size_t)(b * SEQ + s) * EMBED + h * HDIM;
    const float q0 = (float)q[rowQ + lane];
    const float q1 = (float)q[rowQ + lane + 32];

    float m = -1e30f, ssum = 0.f, a0 = 0.f, a1 = 0.f;
    const int j0 = (s - WIN < 0) ? 0 : s - WIN;
    const int j1 = (s + WIN > SEQ - 1) ? SEQ - 1 : s + WIN;

    for (int j = j0; j <= j1; ++j) {
        const size_t rowK = (size_t)(b * SEQ + j) * EMBED + h * HDIM;
        float part = q0 * (float)k[rowK + lane] + q1 * (float)k[rowK + lane + 32];
        #pragma unroll
        for (int off = 16; off; off >>= 1)
            part += __shfl_xor(part, off, 32);
        const float score = part * 0.125f;          // / sqrt(64)

        const float mn   = fmaxf(m, score);
        const float corr = __expf(m - mn);
        const float p    = __expf(score - mn);
        a0   = a0 * corr + p * (float)v[rowK + lane];
        a1   = a1 * corr + p * (float)v[rowK + lane + 32];
        ssum = ssum * corr + p;
        m = mn;
    }
    const float inv = 1.f / ssum;
    attn[rowQ + lane]      = (bf16)(a0 * inv);
    attn[rowQ + lane + 32] = (bf16)(a1 * inv);
}

// ---------------------------------------------------------------------------
extern "C" void kernel_launch(void* const* d_in, const int* in_sizes, int n_in,
                              void* d_out, int out_size, void* d_ws, size_t ws_size,
                              hipStream_t stream) {
    const float* hs = (const float*)d_in[0];
    const float* qw = (const float*)d_in[1];
    const float* qb = (const float*)d_in[2];
    const float* kw = (const float*)d_in[3];
    const float* kb = (const float*)d_in[4];
    const float* vw = (const float*)d_in[5];
    const float* vb = (const float*)d_in[6];
    const float* ow = (const float*)d_in[7];
    const float* ob = (const float*)d_in[8];

    const int M = BATCH * SEQ;   // 4096
    const int N = EMBED;         // 1024
    const int K = EMBED;         // 1024

    char* ws = (char*)d_ws;
    bf16* hsb = (bf16*)ws;  ws += (size_t)M * K * sizeof(bf16);   // 8 MB
    bf16* qwb = (bf16*)ws;  ws += (size_t)N * K * sizeof(bf16);   // 2 MB
    bf16* kwb = (bf16*)ws;  ws += (size_t)N * K * sizeof(bf16);
    bf16* vwb = (bf16*)ws;  ws += (size_t)N * K * sizeof(bf16);
    bf16* owb = (bf16*)ws;  ws += (size_t)N * K * sizeof(bf16);
    bf16* qv  = (bf16*)ws;  ws += (size_t)M * N * sizeof(bf16);   // 8 MB
    bf16* kv  = (bf16*)ws;  ws += (size_t)M * N * sizeof(bf16);
    bf16* vv  = (bf16*)ws;  ws += (size_t)M * N * sizeof(bf16);
    bf16* av  = (bf16*)ws;  ws += (size_t)M * N * sizeof(bf16);

    // fp32 -> bf16 conversions
    {
        int n4 = (M * K) / 4;
        cvt_f32_bf16<<<(n4 + 255) / 256, 256, 0, stream>>>(hs, hsb, n4);
        n4 = (N * K) / 4;
        dim3 g((n4 + 255) / 256);
        cvt_f32_bf16<<<g, 256, 0, stream>>>(qw, qwb, n4);
        cvt_f32_bf16<<<g, 256, 0, stream>>>(kw, kwb, n4);
        cvt_f32_bf16<<<g, 256, 0, stream>>>(vw, vwb, n4);
        cvt_f32_bf16<<<g, 256, 0, stream>>>(ow, owb, n4);
    }

    // Q, K, V projections (bf16 out)
    dim3 gg(N / BN, M / BM);   // (8, 32)
    gemm_bias_wmma<<<gg, 256, 0, stream>>>(hsb, qwb, qb, nullptr, qv, M, N, K);
    gemm_bias_wmma<<<gg, 256, 0, stream>>>(hsb, kwb, kb, nullptr, kv, M, N, K);
    gemm_bias_wmma<<<gg, 256, 0, stream>>>(hsb, vwb, vb, nullptr, vv, M, N, K);

    // sliding-window attention: one wave per (b,h,s)
    {
        int nwaves = BATCH * SEQ * NHEADS;           // 65536
        swin_attn<<<nwaves / 8, 256, 0, stream>>>(qv, kv, vv, av);
    }

    // output projection (fp32 out)
    gemm_bias_wmma<<<gg, 256, 0, stream>>>(av, owb, ob, (float*)d_out, nullptr, M, N, K);
}
